// DA_conv_11115375362102
// MI455X (gfx1250) — matrix operand order repaired and checked by reading, not compiled, not run on hardware
//
#include <hip/hip_runtime.h>
#include <hip/hip_bf16.h>

#define BB 8
#define CC 64
#define HH 128
#define WW 128
#define RR 4
#define HWP (HH*WW)

typedef __attribute__((ext_vector_type(16))) _Float16 v16h;
typedef __attribute__((ext_vector_type(8)))  _Float16 v8h;
typedef __attribute__((ext_vector_type(8)))  float    v8f;

__device__ __forceinline__ float leaky(float x) { return x >= 0.f ? x : 0.1f * x; }

// ---------------------------------------------------------------------------
// Kernel 1: per-sample dynamic depthwise kernels + channel attention + w_conv f16
// ---------------------------------------------------------------------------
__global__ __launch_bounds__(256) void k_dynamic(
    const float* __restrict__ deg,   const float* __restrict__ w_k1,
    const float* __restrict__ w_k2,  const float* __restrict__ w_ca1,
    const float* __restrict__ w_ca2, const float* __restrict__ w_conv,
    float* __restrict__ ker, float* __restrict__ att, _Float16* __restrict__ wf16)
{
    __shared__ float deg_s[CC], y_s[CC], a_s[CC];
    const int b = blockIdx.x, tid = threadIdx.x;
    if (tid < CC) deg_s[tid] = deg[b * CC + tid];
    __syncthreads();
    if (tid < CC) {                       // y = leaky(deg @ w_k1^T)
        float s = 0.f;
        for (int j = 0; j < CC; ++j) s += deg_s[j] * w_k1[tid * CC + j];
        y_s[tid] = leaky(s);
    } else if (tid < 2 * CC) {            // a = leaky(deg @ w_ca1^T)
        const int c = tid - CC; float s = 0.f;
        for (int j = 0; j < CC; ++j) s += deg_s[j] * w_ca1[c * CC + j];
        a_s[c] = leaky(s);
    }
    __syncthreads();
    // kflat[b,g,o] = sum_i y[b,g*16+i] * w_k2[g,o,i]  -> ker[b][c][g][9]
    for (int i = tid; i < RR * 576; i += 256) {
        const int g = i / 576, o = i % 576;
        const float* wp = w_k2 + (g * 576 + o) * 16;
        float s = 0.f;
        #pragma unroll
        for (int j = 0; j < 16; ++j) s += y_s[g * 16 + j] * wp[j];
        const int c = o / 9, t = o % 9;
        ker[((b * CC + c) * RR + g) * 9 + t] = s;
    }
    // att[b,r,c] = sigmoid(sum_i a[b,r*16+i] * w_ca2[r,c,i])
    if (tid < RR * CC) {
        const int r = tid / CC, c = tid % CC;
        const float* wp = w_ca2 + (r * CC + c) * 16;
        float s = 0.f;
        #pragma unroll
        for (int j = 0; j < 16; ++j) s += a_s[r * 16 + j] * wp[j];
        att[(b * RR + r) * CC + c] = 1.f / (1.f + __expf(-s));
    }
    if (b == 0) {                          // f16 copy of w_conv (row-major [o][c])
        for (int i = tid; i < CC * CC; i += 256) wf16[i] = (_Float16)w_conv[i];
    }
}

// ---------------------------------------------------------------------------
// Kernel 2: guide convolutions + first-max argmax routing (r1, r2 bytes)
// ---------------------------------------------------------------------------
__global__ __launch_bounds__(256) void k_route(
    const float* __restrict__ q_map,
    const float* __restrict__ w_g1, const float* __restrict__ b_g1,
    const float* __restrict__ w_g2, const float* __restrict__ b_g2,
    unsigned char* __restrict__ r1buf, unsigned char* __restrict__ r2buf)
{
    __shared__ float wg[2][RR][9];
    __shared__ float bg[2][RR];
    const int tid = threadIdx.x;
    if (tid < RR * 9) { wg[0][tid / 9][tid % 9] = w_g1[tid]; wg[1][tid / 9][tid % 9] = w_g2[tid]; }
    if (tid < RR)     { bg[0][tid] = b_g1[tid]; bg[1][tid] = b_g2[tid]; }
    __syncthreads();
    const int pix = blockIdx.x * 256 + tid;
    const int b = pix / HWP, rem = pix % HWP, h = rem / WW, w = rem % WW;
    const float* qb = q_map + b * HWP;
    float q[9];
    #pragma unroll
    for (int dy = -1; dy <= 1; ++dy)
        #pragma unroll
        for (int dx = -1; dx <= 1; ++dx) {
            const int hh = h + dy, ww = w + dx;
            q[(dy + 1) * 3 + (dx + 1)] =
                (hh >= 0 && hh < HH && ww >= 0 && ww < WW) ? qb[hh * WW + ww] : 0.f;
        }
    int r1 = 0, r2 = 0; float m1 = -1e30f, m2 = -1e30f;
    #pragma unroll
    for (int r = 0; r < RR; ++r) {
        float s1 = bg[0][r], s2 = bg[1][r];
        #pragma unroll
        for (int t = 0; t < 9; ++t) { s1 += q[t] * wg[0][r][t]; s2 += q[t] * wg[1][r][t]; }
        if (s1 > m1) { m1 = s1; r1 = r; }   // strict > keeps first max (argmax semantics)
        if (s2 > m2) { m2 = s2; r2 = r; }
    }
    r1buf[pix] = (unsigned char)r1;
    r2buf[pix] = (unsigned char)r2;
}

// ---------------------------------------------------------------------------
// Kernel 3: fused routed depthwise conv -> f16 LDS -> WMMA 1x1 conv -> epilogue
// One block per (batch, image row). 8 waves of 32 (wave32).
// ---------------------------------------------------------------------------
#define TPAD 72   // 64 + 8 f16 pad: breaks LDS bank conflicts, keeps 16B alignment

__global__ __launch_bounds__(256) void k_main(
    const float* __restrict__ x0,
    const float* __restrict__ ker, const float* __restrict__ att,
    const _Float16* __restrict__ wf16, const float* __restrict__ b_conv,
    const unsigned char* __restrict__ r1buf, const unsigned char* __restrict__ r2buf,
    float* __restrict__ out)
{
    __shared__ _Float16 tmpT[WW][TPAD];   // [pixel][channel], f16
    const int b = blockIdx.x / HH, h = blockIdx.x % HH;
    const int tid = threadIdx.x;
    const unsigned char* r1row = r1buf + b * HWP + h * WW;
    const unsigned char* r2row = r2buf + b * HWP + h * WW;

    // ---- Phase 1: routed dynamic depthwise 3x3 + leaky, store f16 to LDS ----
    #pragma unroll 2
    for (int it = 0; it < (CC * WW) / 256; ++it) {
        const int idx = it * 256 + tid;
        const int c = idx >> 7;         // channel
        const int p = idx & (WW - 1);   // pixel in row (coalesced across lanes)
        const int r = r1row[p];
        const float* kp = ker + ((b * CC + c) * RR + r) * 9;
        const float k0=kp[0],k1=kp[1],k2=kp[2],k3=kp[3],k4=kp[4],k5=kp[5],k6=kp[6],k7=kp[7],k8=kp[8];
        const float* xrow = x0 + ((b * CC + c) * HH + h) * WW;
        const bool pl = (p > 0), pr = (p < WW - 1);
        float s = 0.f;
        if (h > 0) {
            const float* x = xrow - WW;
            if (pl) s += k0 * x[p - 1];
            s += k1 * x[p];
            if (pr) s += k2 * x[p + 1];
        }
        {
            const float* x = xrow;
            if (pl) s += k3 * x[p - 1];
            s += k4 * x[p];
            if (pr) s += k5 * x[p + 1];
        }
        if (h < HH - 1) {
            const float* x = xrow + WW;
            if (pl) s += k6 * x[p - 1];
            s += k7 * x[p];
            if (pr) s += k8 * x[p + 1];
        }
        tmpT[p][c] = (_Float16)leaky(s);
    }
    __syncthreads();

    // ---- Phase 2: D[o,p] = sum_c w_conv[o,c] * tmpT[p,c] via v_wmma_f32_16x16x32_f16
    const int wv = tid >> 5, lane = tid & 31;
    const int mt = wv & 3;            // M-tile: output channels [16*mt, 16*mt+16)
    const int ntb = wv >> 2;          // starting N-tile (pixels), stride 2
    const int m16 = lane & 15, hsel = lane >> 4;

    // A fragments per ISA 16-bit 16x32 layout: lane holds row o=16*mt+m16,
    // halfs 0..7 -> K = 8*hsel+{0..7}, halfs 8..15 -> K = 16+8*hsel+{0..7}
    v16h a0, a1;
    {
        const _Float16* wrow = wf16 + (mt * 16 + m16) * CC;
        const v8h c0 = *(const v8h*)(wrow +  0 + 8 * hsel);
        const v8h c1 = *(const v8h*)(wrow + 16 + 8 * hsel);
        const v8h c2 = *(const v8h*)(wrow + 32 + 8 * hsel);
        const v8h c3 = *(const v8h*)(wrow + 48 + 8 * hsel);
        #pragma unroll
        for (int i = 0; i < 8; ++i) { a0[i] = c0[i]; a0[8 + i] = c1[i];
                                      a1[i] = c2[i]; a1[8 + i] = c3[i]; }
    }

    for (int nt = ntb; nt < 8; nt += 2) {
        const int p = nt * 16 + m16;
        // B fragment per ISA 16-bit 32x16 layout: lane holds column n=p,
        // halfs 0..7 -> K = 8*hsel+{0..7}, halfs 8..15 -> K = 16+8*hsel+{0..7}
        v16h b0, b1;
        {
            const v8h d0 = *(const v8h*)(&tmpT[p][8 * hsel]);
            const v8h d1 = *(const v8h*)(&tmpT[p][16 + 8 * hsel]);
            const v8h d2 = *(const v8h*)(&tmpT[p][32 + 8 * hsel]);
            const v8h d3 = *(const v8h*)(&tmpT[p][48 + 8 * hsel]);
            #pragma unroll
            for (int i = 0; i < 8; ++i) { b0[i] = d0[i]; b0[8 + i] = d1[i];
                                          b1[i] = d2[i]; b1[8 + i] = d3[i]; }
        }
        v8f acc = {};
        acc = __builtin_amdgcn_wmma_f32_16x16x32_f16(false, a0, false, b0,
                                                     (short)0, acc, false, false);
        acc = __builtin_amdgcn_wmma_f32_16x16x32_f16(false, a1, false, b1,
                                                     (short)0, acc, false, false);

        // Epilogue: + b_conv + x0 * att[b, r2, o]  (C/D layout: acc[j] = (M=8*hsel+j, N))
        const int r2 = r2row[p];
        const float* attp = att + (b * RR + r2) * CC;
        #pragma unroll
        for (int j = 0; j < 8; ++j) {
            const int o = mt * 16 + hsel * 8 + j;
            const long ofs = ((long)(b * CC + o) * HH + h) * WW + p;
            out[ofs] = acc[j] + b_conv[o] + x0[ofs] * attp[o];
        }
    }
}

// ---------------------------------------------------------------------------
extern "C" void kernel_launch(void* const* d_in, const int* in_sizes, int n_in,
                              void* d_out, int out_size, void* d_ws, size_t ws_size,
                              hipStream_t stream) {
    (void)in_sizes; (void)n_in; (void)out_size; (void)ws_size;
    const float* x0     = (const float*)d_in[0];
    const float* deg    = (const float*)d_in[1];
    const float* q_map  = (const float*)d_in[2];
    const float* w_k1   = (const float*)d_in[3];
    const float* w_k2   = (const float*)d_in[4];
    const float* w_conv = (const float*)d_in[5];
    const float* b_conv = (const float*)d_in[6];
    const float* w_ca1  = (const float*)d_in[7];
    const float* w_ca2  = (const float*)d_in[8];
    const float* w_g1   = (const float*)d_in[9];
    const float* b_g1   = (const float*)d_in[10];
    const float* w_g2   = (const float*)d_in[11];
    const float* b_g2   = (const float*)d_in[12];

    char* ws = (char*)d_ws;
    size_t off = 0;
    auto carve = [&](size_t bytes) -> void* {
        void* p = ws + off;
        off += (bytes + 255) & ~(size_t)255;
        return p;
    };
    float*        ker  = (float*)carve((size_t)BB * CC * RR * 9 * sizeof(float));
    float*        att  = (float*)carve((size_t)BB * RR * CC * sizeof(float));
    _Float16*     wf16 = (_Float16*)carve((size_t)CC * CC * sizeof(_Float16));
    unsigned char* r1  = (unsigned char*)carve((size_t)BB * HWP);
    unsigned char* r2  = (unsigned char*)carve((size_t)BB * HWP);

    k_dynamic<<<BB, 256, 0, stream>>>(deg, w_k1, w_k2, w_ca1, w_ca2, w_conv,
                                      ker, att, wf16);
    k_route<<<(BB * HWP) / 256, 256, 0, stream>>>(q_map, w_g1, b_g1, w_g2, b_g2, r1, r2);
    k_main<<<BB * HH, 256, 0, stream>>>(x0, ker, att, wf16, b_conv, r1, r2,
                                        (float*)d_out);
}
